// MultiHeadAttention_69483980915344
// MI455X (gfx1250) — matrix operand order repaired and checked by reading, not compiled, not run on hardware
//
#include <hip/hip_runtime.h>
#include <hip/hip_bf16.h>

// ---------------------------------------------------------------------------
// MHA (B=4,S=2048,E=512,H=8,D=64) for gfx1250 / MI455X.
// bf16 WMMA (v_wmma_f32_16x16x32_bf16) + double-buffered TDM
// (tensor_load_to_lds) staging, f32 accumulation, flash-attention.
// ---------------------------------------------------------------------------

typedef __attribute__((ext_vector_type(16))) __bf16        v16bf;
typedef __attribute__((ext_vector_type(8)))  float         v8f;
typedef __attribute__((ext_vector_type(4)))  unsigned int  u32x4;
typedef __attribute__((ext_vector_type(8)))  unsigned int  u32x8;

static constexpr int kB  = 4;
static constexpr int kS  = 2048;
static constexpr int kE  = 512;
static constexpr int kH  = 8;
static constexpr int kD  = 64;
static constexpr int kBS = kB * kS;          // 8192
static constexpr float kScale = 0.125f;      // 1/sqrt(D)

union FragBF { v16bf v; u32x4 q[2]; };
union AccF   { v8f   v; float f[8]; };

static __device__ __forceinline__ unsigned short f2bf(float x) {
  unsigned int u = __builtin_bit_cast(unsigned int, x);
  return (unsigned short)((u + 0x7FFFu + ((u >> 16) & 1u)) >> 16);
}

static __device__ __forceinline__ v8f wmma_bf16(v16bf a, v16bf b, v8f c) {
  // 8 args: (neg_a, A, neg_b, B, c_mod, C, reuse_a, reuse_b)
  return __builtin_amdgcn_wmma_f32_16x16x32_bf16(false, a, false, b,
                                                 (short)0, c, false, false);
}

// Build a 16-element bf16 fragment from two 16-byte chunks.
static __device__ __forceinline__ v16bf ld_frag(const unsigned short* p0,
                                                const unsigned short* p1) {
  FragBF f;
  f.q[0] = *(const u32x4*)p0;
  f.q[1] = *(const u32x4*)p1;
  return f.v;
}

// LDS byte offset of a __shared__ object: generic pointer low 32 bits.
static __device__ __forceinline__ unsigned int lds_off_u32(const void* p) {
  return (unsigned int)(unsigned long long)p;
}

// ---------------------------------------------------------------------------
// Tensor Data Mover: 2-D tile load Global -> LDS with row padding.
// D# group 0/1 per CDNA5 ISA 8.3/8.4.  dims/strides in data_size(=2B) units.
// pad_interval v: pad after 2^(v+1) DWORDs; pad_amount v: (v+1) DWORDs.
// ---------------------------------------------------------------------------
static __device__ __forceinline__ void tdm_load_2d(
    unsigned int lds_addr, const void* gptr,
    unsigned int tensor_d0, unsigned int tensor_d1,
    unsigned long long stride0,
    unsigned int tile_d0, unsigned int tile_d1,
    unsigned int pad_interval, unsigned int pad_amount) {
  unsigned long long ga = (unsigned long long)gptr;
  u32x4 g0;
  g0[0] = 1u;                                    // count=1, user mode
  g0[1] = lds_addr;                              // lds_addr[31:0]
  g0[2] = (unsigned int)(ga & 0xFFFFFFFFu);      // global_addr[31:0]
  g0[3] = (unsigned int)((ga >> 32) & 0x1FFFFFFu) | (2u << 30);  // type=2
  u32x8 g1;
  g1[0] = (1u << 16)                             // data_size = 2 bytes
        | (1u << 20)                             // pad_enable
        | (pad_interval << 22) | (pad_amount << 25);
  g1[1] = (tensor_d0 & 0xFFFFu) << 16;           // atomic_barrier=0 | dim0 lo
  g1[2] = (tensor_d0 >> 16) | ((tensor_d1 & 0xFFFFu) << 16);
  g1[3] = (tensor_d1 >> 16) | (tile_d0 << 16);
  g1[4] = tile_d1 & 0xFFFFu;                     // tile_dim2 = 0
  g1[5] = (unsigned int)(stride0 & 0xFFFFFFFFu); // tensor_dim0_stride lo
  g1[6] = (unsigned int)((stride0 >> 32) & 0xFFFFu);  // hi | dim1_stride=0
  g1[7] = 0u;
  asm volatile("tensor_load_to_lds %0, %1" :: "s"(g0), "s"(g1) : "memory");
}

// ---------------------------------------------------------------------------
// Kernel 1: x (f32) -> xbf (bf16)
// ---------------------------------------------------------------------------
__global__ void k_prep_x(const float* __restrict__ x,
                         unsigned short* __restrict__ xbf, int n4) {
  int i = blockIdx.x * blockDim.x + threadIdx.x;
  if (i >= n4) return;
  float4 v = ((const float4*)x)[i];
  ushort4 o;
  o.x = f2bf(v.x); o.y = f2bf(v.y); o.z = f2bf(v.z); o.w = f2bf(v.w);
  ((ushort4*)xbf)[i] = o;
}

// ---------------------------------------------------------------------------
// Kernel 2: W[k][n] (f32) -> Wt[n][k] (bf16), 4 matrices (Wq,Wk,Wv,Wo)
// ---------------------------------------------------------------------------
__global__ void k_prep_w(const float* __restrict__ w0, const float* __restrict__ w1,
                         const float* __restrict__ w2, const float* __restrict__ w3,
                         unsigned short* __restrict__ wt) {
  __shared__ unsigned short t[32][33];
  const float* w = (blockIdx.z == 0) ? w0 : (blockIdx.z == 1) ? w1
                 : (blockIdx.z == 2) ? w2 : w3;
  unsigned short* o = wt + (size_t)blockIdx.z * kE * kE;
  int c0 = blockIdx.x * 32, r0 = blockIdx.y * 32;
  int tx = threadIdx.x, ty = threadIdx.y;
#pragma unroll
  for (int j = 0; j < 32; j += 8)
    t[ty + j][tx] = f2bf(w[(size_t)(r0 + ty + j) * kE + c0 + tx]);
  __syncthreads();
#pragma unroll
  for (int j = 0; j < 32; j += 8)
    o[(size_t)(c0 + ty + j) * kE + r0 + tx] = t[tx][ty + j];
}

// ---------------------------------------------------------------------------
// Shared GEMM core: C[64 x 128] tile of  A[kBS,kE](bf16) * Wt[N,K]^T(bf16).
// 256 threads = 8 waves; wave (wm,wn) computes rows 16*wm, cols 64*wn.
// X/W tiles double-buffered via the Tensor Data Mover; one barrier/iter:
//   wait(tile i) -> barrier -> issue(tile i+1, other buffer) -> compute(i)
// ---------------------------------------------------------------------------
static constexpr int kXTile = 64 * 40;    // ushorts per X buffer
static constexpr int kWTile = 128 * 40;   // ushorts per W buffer

static __device__ __forceinline__ void gemm_tile(
    const unsigned short* __restrict__ abf,   // [kBS][kE] bf16 row-major
    const unsigned short* __restrict__ wt,    // [kE(N)][kE(K)] bf16 row-major
    int mBase, int nBase,
    unsigned short* ldsx,                     // [2][64][40]
    unsigned short* ldsw,                     // [2][128][40]
    v8f acc[4]) {
  const int tid  = threadIdx.x;
  const int lane = tid & 31;
  const int wave = tid >> 5;
  const int wm   = wave & 3;
  const int wn   = wave >> 2;
  const int half = lane >> 4;
  const int l16  = lane & 15;
  const unsigned int ldsx_base = lds_off_u32(ldsx);
  const unsigned int ldsw_base = lds_off_u32(ldsw);
  const int nIter = kE / 32;

  if (wave == 0) {   // prologue: tile 0 -> buffer 0
    tdm_load_2d(ldsx_base, abf + (size_t)mBase * kE,
                kE, (unsigned)(kBS - mBase), kE, 32, 64, 3, 3);
    tdm_load_2d(ldsw_base, wt + (size_t)nBase * kE,
                kE, (unsigned)(kE - nBase), kE, 32, 128, 3, 3);
  }

  for (int it = 0; it < nIter; ++it) {
    const int cur = it & 1;
    if (wave == 0) __builtin_amdgcn_s_wait_tensorcnt(0);
    __syncthreads();                     // publish tile it; retire old reads
    if (wave == 0 && it + 1 < nIter) {   // overlap: copy tile it+1
      const int kk = (it + 1) * 32;
      tdm_load_2d(ldsx_base + (unsigned)((cur ^ 1) * kXTile * 2),
                  abf + (size_t)mBase * kE + kk,
                  kE, (unsigned)(kBS - mBase), kE, 32, 64, 3, 3);
      tdm_load_2d(ldsw_base + (unsigned)((cur ^ 1) * kWTile * 2),
                  wt + (size_t)nBase * kE + kk,
                  kE, (unsigned)(kE - nBase), kE, 32, 128, 3, 3);
    }
    const unsigned short* xb = ldsx + cur * kXTile;
    const unsigned short* wb = ldsw + cur * kWTile;

    // A fragment: row = wm*16 + l16, k chunks at half*8 and half*8+16
    const unsigned short* ap = xb + (wm * 16 + l16) * 40 + half * 8;
    v16bf a = ld_frag(ap, ap + 16);
#pragma unroll
    for (int nt = 0; nt < 4; ++nt) {
      // B fragment: col n = wn*64 + nt*16 + l16, k chunks at half*8 and half*8+16
      const unsigned short* bp = wb + (wn * 64 + nt * 16 + l16) * 40 + half * 8;
      v16bf b = ld_frag(bp, bp + 16);
      acc[nt] = wmma_bf16(a, b, acc[nt]);
    }
  }
}

// ---------------------------------------------------------------------------
// Kernel 3: QKV projections -> head-split bf16 [3][B*H, S, D] (+bias)
// ---------------------------------------------------------------------------
__global__ void __launch_bounds__(256) k_gemm_qkv(
    const unsigned short* __restrict__ xbf,
    const unsigned short* __restrict__ wtAll,   // [3][kE][kE]
    const float* __restrict__ bq, const float* __restrict__ bk,
    const float* __restrict__ bv,
    unsigned short* __restrict__ qkv) {
  __shared__ unsigned short ldsx[2 * kXTile];
  __shared__ unsigned short ldsw[2 * kWTile];
  const int z = blockIdx.z;
  const unsigned short* wt = wtAll + (size_t)z * kE * kE;
  const float* bias = (z == 0) ? bq : (z == 1) ? bk : bv;
  unsigned short* dst = qkv + (size_t)z * kBS * kE;
  const int mBase = blockIdx.y * 64;
  const int nBase = blockIdx.x * 128;

  v8f acc[4] = {};
  gemm_tile(xbf, wt, mBase, nBase, ldsx, ldsw, acc);

  const int lane = threadIdx.x & 31;
  const int wave = threadIdx.x >> 5;
  const int wm = wave & 3, wn = wave >> 2;
  const int half = lane >> 4, l16 = lane & 15;
#pragma unroll
  for (int nt = 0; nt < 4; ++nt) {
    const int e = nBase + wn * 64 + nt * 16 + l16;
    const int h = e >> 6, d = e & 63;
    const float bb = bias[e];
    AccF a; a.v = acc[nt];
#pragma unroll
    for (int v = 0; v < 8; ++v) {
      int rowg = mBase + wm * 16 + half * 8 + v;
      int b = rowg >> 11;          // / kS
      int s = rowg & (kS - 1);
      dst[((size_t)(b * kH + h) * kS + s) * kD + d] = f2bf(a.f[v] + bb);
    }
  }
}

// ---------------------------------------------------------------------------
// Kernel 4: flash attention.  Grid: (S/128, B*H), 256 threads = 8 waves.
// Each wave: 16 query rows, full D=64 accumulators, 32-key blocks.
// K tile double-buffered via TDM; V double-buffered manually (transposed).
// ---------------------------------------------------------------------------
static constexpr int kKTile = 32 * 72;    // ushorts per K buffer
static constexpr int kVTile = 64 * 40;    // ushorts per V buffer

__global__ void __launch_bounds__(256) k_flash(
    const unsigned short* __restrict__ qbuf,  // [B*H, S, D] bf16
    const unsigned short* __restrict__ kbuf,
    const unsigned short* __restrict__ vbuf,
    unsigned short* __restrict__ attnbf) {    // [kBS, kE] bf16 (merged)
  __shared__ unsigned short ldsk[2 * kKTile];       // K rows [key][d]
  __shared__ unsigned short ldsv[2 * kVTile];       // V transposed [d][key]
  __shared__ unsigned short ldsp[8 * 16 * 40];      // per-wave P staging

  const int tid  = threadIdx.x;
  const int lane = tid & 31;
  const int wave = tid >> 5;
  const int half = lane >> 4, l16 = lane & 15;
  const int bh   = blockIdx.y;
  const int q0   = blockIdx.x * 128 + wave * 16;

  const unsigned short* qp = qbuf + (size_t)bh * kS * kD;
  const unsigned short* kp = kbuf + (size_t)bh * kS * kD;
  const unsigned short* vp = vbuf + (size_t)bh * kS * kD;
  const unsigned int ldsk_base = lds_off_u32(ldsk);

  // Q fragments: two 16x32 d-halves (A layout, straight from global)
  const unsigned short* qrow = qp + (size_t)(q0 + l16) * kD + half * 8;
  v16bf qa0 = ld_frag(qrow, qrow + 16);
  v16bf qa1 = ld_frag(qrow + 32, qrow + 48);

  v8f acc[4] = {};
  float mrow[8], lrow[8];
#pragma unroll
  for (int v = 0; v < 8; ++v) { mrow[v] = -3.4e38f; lrow[v] = 0.f; }

  unsigned short* pme = ldsp + wave * 16 * 40;
  const int nIter = kS / 32;

  if (wave == 0) {   // prologue: K tile 0 -> buffer 0
    tdm_load_2d(ldsk_base, kp, kD, kS, kD, kD, 32, 4, 3);
  }

  for (int it = 0; it < nIter; ++it) {
    const int cur = it & 1;
    const int kb = it * 32;
    {  // stage V transposed into ldsv[cur]: 8 bf16 per thread (256 thr)
      unsigned short* vdst = ldsv + cur * kVTile;
      int row = tid >> 3;          // 0..31
      int db  = (tid & 7) * 8;     // 0..56
      const unsigned short* vs = vp + (size_t)(kb + row) * kD + db;
      u32x4 e0 = *(const u32x4*)vs;
      const unsigned short* p0 = (const unsigned short*)&e0;
#pragma unroll
      for (int j = 0; j < 8; ++j) vdst[(db + j) * 40 + row] = p0[j];
    }
    if (kb + 32 < kS)
      __builtin_prefetch(vp + (size_t)(kb + 32 + (tid >> 3)) * kD, 0, 0);
    if (wave == 0) __builtin_amdgcn_s_wait_tensorcnt(0);
    __syncthreads();                     // publish K/V tile it
    if (wave == 0 && it + 1 < nIter) {   // overlap: K tile it+1
      tdm_load_2d(ldsk_base + (unsigned)((cur ^ 1) * kKTile * 2),
                  kp + (size_t)(kb + 32) * kD,
                  kD, (unsigned)(kS - kb - 32), kD, kD, 32, 4, 3);
    }
    const unsigned short* kbase = ldsk + cur * kKTile;
    const unsigned short* vbase = ldsv + cur * kVTile;

    // ---- scores: two 16x16 key tiles, K-dim = 64 (2 WMMAs each) ----
    v8f s01[2];
#pragma unroll
    for (int kt = 0; kt < 2; ++kt) {
      const unsigned short* kr = kbase + (kt * 16 + l16) * 72 + half * 8;
      v16bf b0 = ld_frag(kr, kr + 16);           // d = 0..31
      v16bf b1 = ld_frag(kr + 32, kr + 48);      // d = 32..63
      v8f s = {};
      s = wmma_bf16(qa0, b0, s);
      s = wmma_bf16(qa1, b1, s);
      s01[kt] = s;
    }

    // ---- online softmax (row stats replicated across 16-lane halves) ----
    AccF t0; t0.v = s01[0];
    AccF t1; t1.v = s01[1];
    float alpha[8];
#pragma unroll
    for (int v = 0; v < 8; ++v) {
      float a0 = t0.f[v] * kScale;
      float a1 = t1.f[v] * kScale;
      float c = fmaxf(a0, a1);
      c = fmaxf(c, __shfl_xor(c, 1, 32));
      c = fmaxf(c, __shfl_xor(c, 2, 32));
      c = fmaxf(c, __shfl_xor(c, 4, 32));
      c = fmaxf(c, __shfl_xor(c, 8, 32));
      float mnew = fmaxf(mrow[v], c);
      float al = __expf(mrow[v] - mnew);
      float p0 = __expf(a0 - mnew);
      float p1 = __expf(a1 - mnew);
      float rs = p0 + p1;
      rs += __shfl_xor(rs, 1, 32);
      rs += __shfl_xor(rs, 2, 32);
      rs += __shfl_xor(rs, 4, 32);
      rs += __shfl_xor(rs, 8, 32);
      lrow[v] = lrow[v] * al + rs;
      mrow[v] = mnew;
      alpha[v] = al;
      int prow = half * 8 + v;                 // C-layout row -> LDS row
      pme[prow * 40 + l16]      = f2bf(p0);
      pme[prow * 40 + 16 + l16] = f2bf(p1);
    }
#pragma unroll
    for (int dt = 0; dt < 4; ++dt) {
      AccF a; a.v = acc[dt];
#pragma unroll
      for (int v = 0; v < 8; ++v) a.f[v] *= alpha[v];
      acc[dt] = a.v;
    }

    // ---- P(16x32) x V(32x64): A-frag from LDS P staging ----
    const unsigned short* pr = pme + l16 * 40 + half * 8;
    v16bf pa = ld_frag(pr, pr + 16);
#pragma unroll
    for (int dt = 0; dt < 4; ++dt) {
      const unsigned short* vr = vbase + (dt * 16 + l16) * 40 + half * 8;
      v16bf vb = ld_frag(vr, vr + 16);
      acc[dt] = wmma_bf16(pa, vb, acc[dt]);
    }
  }

  // ---- normalize + store merged [B,S,E] bf16 ----
  const int b = bh >> 3, h = bh & 7;
#pragma unroll
  for (int dt = 0; dt < 4; ++dt) {
    AccF a; a.v = acc[dt];
#pragma unroll
    for (int v = 0; v < 8; ++v) {
      int row = half * 8 + v;
      float inv = 1.f / lrow[v];
      int bs = b * kS + q0 + row;
      int e  = h * 64 + dt * 16 + l16;
      attnbf[(size_t)bs * kE + e] = f2bf(a.f[v] * inv);
    }
  }
}

// ---------------------------------------------------------------------------
// Kernel 5: output projection -> f32 d_out (+bias)
// ---------------------------------------------------------------------------
__global__ void __launch_bounds__(256) k_gemm_out(
    const unsigned short* __restrict__ attnbf,
    const unsigned short* __restrict__ wot,
    const float* __restrict__ bo,
    float* __restrict__ out) {
  __shared__ unsigned short ldsx[2 * kXTile];
  __shared__ unsigned short ldsw[2 * kWTile];
  const int mBase = blockIdx.y * 64;
  const int nBase = blockIdx.x * 128;

  v8f acc[4] = {};
  gemm_tile(attnbf, wot, mBase, nBase, ldsx, ldsw, acc);

  const int lane = threadIdx.x & 31;
  const int wave = threadIdx.x >> 5;
  const int wm = wave & 3, wn = wave >> 2;
  const int half = lane >> 4, l16 = lane & 15;
#pragma unroll
  for (int nt = 0; nt < 4; ++nt) {
    const int e = nBase + wn * 64 + nt * 16 + l16;
    const float bb = bo[e];
    AccF a; a.v = acc[nt];
#pragma unroll
    for (int v = 0; v < 8; ++v) {
      int rowg = mBase + wm * 16 + half * 8 + v;
      out[(size_t)rowg * kE + e] = a.f[v] + bb;
    }
  }
}

// ---------------------------------------------------------------------------
extern "C" void kernel_launch(void* const* d_in, const int* in_sizes, int n_in,
                              void* d_out, int out_size, void* d_ws, size_t ws_size,
                              hipStream_t stream) {
  (void)in_sizes; (void)n_in; (void)out_size; (void)ws_size;
  const float* x  = (const float*)d_in[0];
  const float* Wq = (const float*)d_in[1];
  const float* bq = (const float*)d_in[2];
  const float* Wk = (const float*)d_in[3];
  const float* bk = (const float*)d_in[4];
  const float* Wv = (const float*)d_in[5];
  const float* bv = (const float*)d_in[6];
  const float* Wo = (const float*)d_in[7];
  const float* bo = (const float*)d_in[8];
  float* out = (float*)d_out;

  // Workspace layout (bytes):
  //   [0,      8 MiB)  xbf   : x in bf16           (kBS*kE)
  //   [8 MiB, 10 MiB)  wt    : 4 transposed bf16 W (4*kE*kE)
  //   [10 MiB,34 MiB)  qkv   : Q,K,V bf16 head-split [3][B*H,S,D]
  //   [34 MiB,42 MiB)  attn  : merged attention bf16 [kBS,kE]
  char* ws = (char*)d_ws;
  unsigned short* xbf  = (unsigned short*)(ws);
  unsigned short* wt   = (unsigned short*)(ws + (size_t)8388608);
  unsigned short* qkv  = (unsigned short*)(ws + (size_t)10485760);
  unsigned short* attn = (unsigned short*)(ws + (size_t)35651584);

  const size_t nElem = (size_t)kBS * kE;          // 4,194,304
  const size_t wElem = (size_t)kE * kE;           // 262,144

  k_prep_x<<<(int)(nElem / 4 / 256), 256, 0, stream>>>(x, xbf, (int)(nElem / 4));
  k_prep_w<<<dim3(16, 16, 4), dim3(32, 8), 0, stream>>>(Wq, Wk, Wv, Wo, wt);
  k_gemm_qkv<<<dim3(kE / 128, kBS / 64, 3), 256, 0, stream>>>(
      xbf, wt, bq, bk, bv, qkv);
  k_flash<<<dim3(kS / 128, kB * kH), 256, 0, stream>>>(
      qkv, qkv + nElem, qkv + 2 * nElem, attn);
  k_gemm_out<<<dim3(kE / 128, kBS / 64), 256, 0, stream>>>(
      attn, wt + 3 * wElem, bo, out);
}
